// RNN_49297634623576
// MI455X (gfx1250) — matrix operand order repaired and checked by reading, not compiled, not run on hardware
//
#include <hip/hip_runtime.h>
#include <hip/hip_bf16.h>

typedef __attribute__((ext_vector_type(16))) _Float16 v16h;
typedef __attribute__((ext_vector_type(8)))  _Float16 v8h;
typedef __attribute__((ext_vector_type(8)))  float    v8f;
typedef __attribute__((ext_vector_type(4)))  float    v4f;

__device__ __forceinline__ float fast_tanh(float v) {
#if __has_builtin(__builtin_amdgcn_tanhf)
    return __builtin_amdgcn_tanhf(v);          // v_tanh_f32 (gfx1250 TRANS op)
#elif __has_builtin(__builtin_amdgcn_tanh_f32)
    return __builtin_amdgcn_tanh_f32(v);
#else
    return tanhf(v);
#endif
}

// One wave handles 16 batch rows.
//   GEMM1: (16x28 pad 32) x (28x64)  -> 4x v_wmma_f32_16x16x32_f16
//   tanh + bias in registers, h staged f16 in wave-private LDS tile
//   GEMM2: (16x64) x (64x10 pad 16)  -> 2x v_wmma_f32_16x16x32_f16
__global__ __launch_bounds__(256)
void rnn_fused_wmma(const float* __restrict__ x,
                    const float* __restrict__ w_ih,
                    const float* __restrict__ b_ih,
                    const float* __restrict__ b_hh,
                    const float* __restrict__ fc_w,
                    const float* __restrict__ fc_b,
                    float* __restrict__ out,
                    int Btot)
{
    // Weight fragments in WMMA B-layout (K striped across lanes within a VGPR):
    // element i of lane L  ->  K = (i&7) + 8*(L>>4) + 16*(i>>3) (+32*kt),  N = (L&15) (+16*nt)
    __shared__ v16h  sB1[4][32];   // w_ih^T  (28x64, K-padded), 4 N-tiles
    __shared__ v16h  sB2[2][32];   // fc_w^T  (64x10, N-padded), 2 K-tiles
    __shared__ float sBias[64];    // b_ih + b_hh
    __shared__ float sFcb[16];
    __shared__ _Float16 __attribute__((aligned(32))) hbuf[8][16][64]; // per-wave h tile

    const int t = threadIdx.x;

    // ---- Build weight fragments once per block (tiny: ~6 KB) ----
    {
        _Float16* p = (_Float16*)sB1;                 // 2048 halfs
        #pragma unroll
        for (int u = 0; u < 8; ++u) {
            int e    = t * 8 + u;
            int nt   = e >> 9, lane = (e >> 4) & 31, i = e & 15;
            int K    = (i & 7) + ((lane >> 4) << 3) + ((i >> 3) << 4);
            int N    = (nt << 4) + (lane & 15);
            p[e] = (_Float16)((K < 28) ? w_ih[N * 28 + K] : 0.0f);
        }
        _Float16* q = (_Float16*)sB2;                 // 1024 halfs
        #pragma unroll
        for (int u = 0; u < 4; ++u) {
            int e    = t * 4 + u;
            int kt   = e >> 9, lane = (e >> 4) & 31, i = e & 15;
            int K    = (kt << 5) + ((lane >> 4) << 3) + ((i >> 3) << 4) + (i & 7);
            int N    = lane & 15;
            q[e] = (_Float16)((N < 10) ? fc_w[N * 64 + K] : 0.0f);
        }
        if (t < 64)             sBias[t]     = b_ih[t] + b_hh[t];
        if (t >= 64 && t < 80)  sFcb[t - 64] = ((t - 64) < 10) ? fc_b[t - 64] : 0.0f;
    }
    __syncthreads();

    const int  lane = t & 31;
    const int  wave = t >> 5;
    const int  hi   = lane >> 4;      // half-wave select
    const int  ln   = lane & 15;
    const long row0 = (((long)blockIdx.x * 8 + wave) << 4);
    const long row  = row0 + ln;
    const long rl   = (row < Btot) ? row : (long)(Btot - 1);

    // ---- A fragment: x[row][27][0..27], zero-padded K=28..31 ----
    // A layout (16-bit 16x32): lane ln/hi holds K in {8*hi..8*hi+7} U {16+8*hi..23+8*hi}
    // -> two contiguous 8-float runs: perfect for float4-pair loads.
    const float* xr = x + rl * 784 + 756;             // 27*28 = 756
    const v4f* pa = (const v4f*)(xr + (hi << 3));
    v4f c0 = pa[0];
    v4f c1 = pa[1];
    const v4f* pb = (const v4f*)(xr + 16 + (hi << 3));
    v4f c2 = pb[0];
    v4f c3 = {0.0f, 0.0f, 0.0f, 0.0f};
    if (hi == 0) c3 = pb[1];                          // hi=1: K=28..31 are pad -> 0

    v16h a;
    a[0]=(_Float16)c0.x;  a[1]=(_Float16)c0.y;  a[2]=(_Float16)c0.z;  a[3]=(_Float16)c0.w;
    a[4]=(_Float16)c1.x;  a[5]=(_Float16)c1.y;  a[6]=(_Float16)c1.z;  a[7]=(_Float16)c1.w;
    a[8]=(_Float16)c2.x;  a[9]=(_Float16)c2.y;  a[10]=(_Float16)c2.z; a[11]=(_Float16)c2.w;
    a[12]=(_Float16)c3.x; a[13]=(_Float16)c3.y; a[14]=(_Float16)c3.z; a[15]=(_Float16)c3.w;

    // ---- GEMM1: h = tanh(x @ w_ih^T + bias), 4 N-tiles ----
    #pragma unroll
    for (int nt = 0; nt < 4; ++nt) {
        v8f acc = {};
        acc = __builtin_amdgcn_wmma_f32_16x16x32_f16(
                  false, a, false, sB1[nt][lane], (short)0, acc, false, false);
        float bn = sBias[(nt << 4) + ln];
        // D layout: VGPR r -> M = r + 8*hi, N = ln (+16*nt)
        #pragma unroll
        for (int r = 0; r < 8; ++r) {
            hbuf[wave][r + (hi << 3)][(nt << 4) + ln] =
                (_Float16)fast_tanh(acc[r] + bn);
        }
    }
    __syncthreads();

    // ---- GEMM2: out = h @ fc_w^T + fc_b, 2 K-tiles ----
    v8f acc2 = {};
    #pragma unroll
    for (int kt = 0; kt < 2; ++kt) {
        // Re-read h in A-layout: two contiguous 8-half runs per lane (b128 LDS loads)
        v8h lo = *(const v8h*)&hbuf[wave][ln][(kt << 5) + (hi << 3)];
        v8h hh = *(const v8h*)&hbuf[wave][ln][(kt << 5) + 16 + (hi << 3)];
        v16h a2 = __builtin_shufflevector(lo, hh,
                      0,1,2,3,4,5,6,7,8,9,10,11,12,13,14,15);
        acc2 = __builtin_amdgcn_wmma_f32_16x16x32_f16(
                   false, a2, false, sB2[kt][lane], (short)0, acc2, false, false);
    }

    if (ln < 10) {
        float fb = sFcb[ln];
        #pragma unroll
        for (int r = 0; r < 8; ++r) {
            long orow = row0 + r + (hi << 3);
            if (orow < Btot) out[orow * 10 + ln] = acc2[r] + fb;
        }
    }
}

extern "C" void kernel_launch(void* const* d_in, const int* in_sizes, int n_in,
                              void* d_out, int out_size, void* d_ws, size_t ws_size,
                              hipStream_t stream) {
    const float* x    = (const float*)d_in[0];
    const float* w_ih = (const float*)d_in[1];
    // d_in[2] = w_hh : unused, hx == 0 so the recurrent term vanishes
    const float* b_ih = (const float*)d_in[3];
    const float* b_hh = (const float*)d_in[4];
    const float* fc_w = (const float*)d_in[5];
    const float* fc_b = (const float*)d_in[6];
    float* out = (float*)d_out;

    const int B = in_sizes[0] / (28 * 28);            // 65536
    const int nblocks = (B + 127) / 128;              // 128 rows per 256-thread block
    rnn_fused_wmma<<<nblocks, 256, 0, stream>>>(x, w_ih, b_ih, b_hh, fc_w, fc_b, out, B);
}
